// RAGCRN_21483426414778
// MI455X (gfx1250) — matrix operand order repaired and checked by reading, not compiled, not run on hardware
//
#include <hip/hip_runtime.h>
#include <stdint.h>

#define HIDD   64
#define SEQN   12
#define NPREDN 12
#define TOPKN  9
#define BATCH  32
#define NODES  1024
#define ROWS   (BATCH*NODES)     /* 32768 */
#define F1     (HIDD+1)          /* 65   */
#define KPAD   96
#define COLS   (BATCH*F1)        /* 2080 */
#define RH     (ROWS*HIDD)       /* 2097152 */

typedef __attribute__((ext_vector_type(16))) __bf16 v16bf;
typedef __attribute__((ext_vector_type(8)))  float  v8f;

union Frag { v16bf v; uint4 u[2]; };

__device__ __forceinline__ uint16_t f2bf(float x) {
  union { float f; uint32_t u; } v; v.f = x;
  uint32_t r = v.u + 0x7FFFu + ((v.u >> 16) & 1u);
  return (uint16_t)(r >> 16);
}
__device__ __forceinline__ float bf2f(uint16_t h) {
  union { uint32_t u; float f; } v; v.u = ((uint32_t)h) << 16;
  return v.f;
}
__device__ __forceinline__ float sigf(float x) { return 1.f / (1.f + expf(-x)); }

// A fragment (16x32 bf16): lane = row(0..15) | khalf-sel; p = Arow + kt + kh
__device__ __forceinline__ v16bf load_a(const uint16_t* p) {
  Frag f; f.u[0] = *(const uint4*)p; f.u[1] = *(const uint4*)(p + 16); return f.v;
}
// B fragment (32x16 bf16 staged [N][K]): lane = col | koff-sel; p = Bcol + kt (+ko)
__device__ __forceinline__ v16bf load_b(const uint16_t* p) {
  Frag f; f.u[0] = *(const uint4*)p; f.u[1] = *(const uint4*)(p + 16); return f.v;
}

// ---------------- adjacency normalization ----------------
__global__ void k_deg(const float* __restrict__ adj, float* __restrict__ dinv) {
  int lane = threadIdx.x & 31;
  int row  = blockIdx.x * 8 + (threadIdx.x >> 5);
  if (row >= NODES) return;
  float s = 0.f;
  for (int c = lane; c < NODES; c += 32) s += adj[row * NODES + c];
  for (int o = 16; o; o >>= 1) s += __shfl_xor(s, o);
  if (lane == 0) {
    float d = s + 1.0f;
    dinv[row] = d > 0.f ? rsqrtf(d) : 0.f;
  }
}

// a_hat[m,n] = dinv[m] * (adj[n,m] + I) * dinv[n], stored bf16 row-major [M=1024,K=1024]
__global__ void k_ahat(const float* __restrict__ adj, const float* __restrict__ dinv,
                       uint16_t* __restrict__ ah) {
  int tid = blockIdx.x * blockDim.x + threadIdx.x;
  if (tid >= NODES * NODES) return;
  int m = tid / NODES, n = tid % NODES;
  float v = dinv[m] * (adj[n * NODES + m] + (m == n ? 1.f : 0.f)) * dinv[n];
  ah[tid] = f2bf(v);
}

// stage weight into Bt[N][Kp] bf16 (K-padded with zeros)
__global__ void k_stage_w(uint16_t* __restrict__ dst, const float* __restrict__ src,
                          int N, int Kreal, int Kp, int rs, int cs) {
  int tid = blockIdx.x * blockDim.x + threadIdx.x;
  if (tid >= N * Kp) return;
  int j = tid / Kp, k = tid % Kp;
  dst[tid] = f2bf(k < Kreal ? src[j * rs + k * cs] : 0.f);
}

// Build CC^T bf16 [col=b*65+f][k=n]; mode1 applies the (torch-flat) reset gate r*h
__global__ void k_build_cc(const float* __restrict__ train, const float* __restrict__ h,
                           const float* __restrict__ sigG, uint16_t* __restrict__ cct,
                           int t, int mode) {
  int tid = blockIdx.x * blockDim.x + threadIdx.x;
  if (tid >= COLS * NODES) return;
  int n = tid % NODES, col = tid / NODES;
  int b = col / F1, f = col % F1;
  float v;
  if (f == 0) {
    v = train[(b * NODES + n) * SEQN + t];
  } else {
    int fh = f - 1;
    float hv = h[(b * NODES + n) * HIDD + fh];
    if (mode) {
      int m = n * HIDD + fh;                       // flat [B, N*H] index
      float r = sigG[(b * NODES + (m >> 7)) * 128 + (m & 127)];
      hv = r * hv;
    }
    v = hv;
  }
  cct[col * NODES + n] = f2bf(v);
}

// scatter-store one 16x16 f32 accumulator tile -> AC_bf16[(b*N+m)*KPAD + f]
__device__ __forceinline__ void store_ahat_tile(v8f acc, uint16_t* __restrict__ AC,
                                                int mrow, int ncol, int lane) {
  int col = ncol + (lane & 15);
  int b = col / F1, f = col % F1;
  int mbase = mrow + 8 * (lane >> 4);
#pragma unroll
  for (int v = 0; v < 8; ++v)
    AC[(size_t)(b * NODES + mbase + v) * KPAD + f] = f2bf(acc[v]);
}

// GEMM1: AC = a_hat[1024x1024] @ CC[1024 x 2080]; 2x2 WMMA tiles per wave
__global__ void k_gemm_ahat(const uint16_t* __restrict__ A, const uint16_t* __restrict__ Bt,
                            uint16_t* __restrict__ AC) {
  int lane = threadIdx.x & 31;
  int wid  = blockIdx.x * (blockDim.x >> 5) + (threadIdx.x >> 5);
  const int NB = COLS / 32;                         // 65 n-blocks
  if (wid >= (NODES / 32) * NB) return;             // 32*65 = 2080 waves
  int mb = wid / NB, nb = wid % NB;
  const uint16_t* ap0 = A  + (size_t)(mb * 32 + (lane & 15)) * NODES + ((lane >> 4) * 8);
  const uint16_t* ap1 = ap0 + (size_t)16 * NODES;
  const uint16_t* bp0 = Bt + (size_t)(nb * 32 + (lane & 15)) * NODES + ((lane >> 4) * 8);
  const uint16_t* bp1 = bp0 + (size_t)16 * NODES;
  v8f a00 = {}, a01 = {}, a10 = {}, a11 = {};
  for (int kt = 0; kt < NODES; kt += 32) {
    __builtin_prefetch(ap0 + kt + 512, 0, 1);
    __builtin_prefetch(ap1 + kt + 512, 0, 1);
    __builtin_prefetch(bp0 + kt + 512, 0, 1);
    __builtin_prefetch(bp1 + kt + 512, 0, 1);
    v16bf fa0 = load_a(ap0 + kt), fa1 = load_a(ap1 + kt);
    v16bf fb0 = load_b(bp0 + kt), fb1 = load_b(bp1 + kt);
    a00 = __builtin_amdgcn_wmma_f32_16x16x32_bf16(false, fa0, false, fb0, (short)0, a00, false, false);
    a01 = __builtin_amdgcn_wmma_f32_16x16x32_bf16(false, fa0, false, fb1, (short)0, a01, false, false);
    a10 = __builtin_amdgcn_wmma_f32_16x16x32_bf16(false, fa1, false, fb0, (short)0, a10, false, false);
    a11 = __builtin_amdgcn_wmma_f32_16x16x32_bf16(false, fa1, false, fb1, (short)0, a11, false, false);
  }
  store_ahat_tile(a00, AC, mb * 32,      nb * 32,      lane);
  store_ahat_tile(a01, AC, mb * 32,      nb * 32 + 16, lane);
  store_ahat_tile(a10, AC, mb * 32 + 16, nb * 32,      lane);
  store_ahat_tile(a11, AC, mb * 32 + 16, nb * 32 + 16, lane);
}

// store one 16x16 f32 tile with bias + activation; optional f32 and bf16 outputs
__device__ __forceinline__ void store_rm_tile(v8f acc, float* __restrict__ Cf,
                                              uint16_t* __restrict__ Cbf,
                                              const float* __restrict__ bias,
                                              int N, int act, int mrow, int ncol, int lane) {
  int n = ncol + (lane & 15);
  float bv = bias ? bias[n] : 0.f;
  int mbase = mrow + 8 * (lane >> 4);
#pragma unroll
  for (int v = 0; v < 8; ++v) {
    float val = acc[v] + bv;
    if (act == 1) val = sigf(val);
    else if (act == 2) val = tanhf(val);
    size_t o = (size_t)(mbase + v) * N + n;
    if (Cf)  Cf[o]  = val;
    if (Cbf) Cbf[o] = f2bf(val);
  }
}

// General GEMM: C[M,N] = A_bf16[M,K] @ Bt_bf16[N,K]^T + bias; 2x2 WMMA tiles per wave
// Requires M%32==0, N%32==0, K%32==0.
__global__ void k_gemm_rm(const uint16_t* __restrict__ A, const uint16_t* __restrict__ Bt,
                          const float* __restrict__ bias, float* __restrict__ Cf,
                          uint16_t* __restrict__ Cbf, int M, int N, int K, int act) {
  int lane = threadIdx.x & 31;
  int wid  = blockIdx.x * (blockDim.x >> 5) + (threadIdx.x >> 5);
  int NB = N >> 5;
  if (wid >= (M >> 5) * NB) return;
  int mb = wid / NB, nb = wid % NB;
  const uint16_t* ap0 = A  + (size_t)(mb * 32 + (lane & 15)) * K + ((lane >> 4) * 8);
  const uint16_t* ap1 = ap0 + (size_t)16 * K;
  const uint16_t* bp0 = Bt + (size_t)(nb * 32 + (lane & 15)) * K + ((lane >> 4) * 8);
  const uint16_t* bp1 = bp0 + (size_t)16 * K;
  v8f a00 = {}, a01 = {}, a10 = {}, a11 = {};
  for (int kt = 0; kt < K; kt += 32) {
    v16bf fa0 = load_a(ap0 + kt), fa1 = load_a(ap1 + kt);
    v16bf fb0 = load_b(bp0 + kt), fb1 = load_b(bp1 + kt);
    a00 = __builtin_amdgcn_wmma_f32_16x16x32_bf16(false, fa0, false, fb0, (short)0, a00, false, false);
    a01 = __builtin_amdgcn_wmma_f32_16x16x32_bf16(false, fa0, false, fb1, (short)0, a01, false, false);
    a10 = __builtin_amdgcn_wmma_f32_16x16x32_bf16(false, fa1, false, fb0, (short)0, a10, false, false);
    a11 = __builtin_amdgcn_wmma_f32_16x16x32_bf16(false, fa1, false, fb1, (short)0, a11, false, false);
  }
  store_rm_tile(a00, Cf, Cbf, bias, N, act, mb * 32,      nb * 32,      lane);
  store_rm_tile(a01, Cf, Cbf, bias, N, act, mb * 32,      nb * 32 + 16, lane);
  store_rm_tile(a10, Cf, Cbf, bias, N, act, mb * 32 + 16, nb * 32,      lane);
  store_rm_tile(a11, Cf, Cbf, bias, N, act, mb * 32 + 16, nb * 32 + 16, lane);
}

// h_new = u*h + (1-u)*c with torch-flat chunked u
__global__ void k_update_h(const float* __restrict__ hprev, const float* __restrict__ sigG,
                           const float* __restrict__ c2, float* __restrict__ hout) {
  int tid = blockIdx.x * blockDim.x + threadIdx.x;
  if (tid >= RH) return;
  int row = tid >> 6, f = tid & 63;
  int b = row >> 10, node = row & 1023;
  int m = node * HIDD + f;
  float u = sigG[((b << 10) + 512 + (m >> 7)) * 128 + (m & 127)];
  hout[tid] = u * hprev[tid] + (1.f - u) * c2[tid];
}

// AttentionMode scoring: result[t] = sum_rows (gelu(e@wm1^T+bm1)@wm2^T + bm2)
__global__ void k_score(const float* __restrict__ enc, const float* __restrict__ wm1,
                        const float* __restrict__ bm1, const float* __restrict__ wm2,
                        const float* __restrict__ bm2, float* __restrict__ result) {
  __shared__ float red[256];
  int t   = blockIdx.x >> 7;                        // 128 blocks per timestep
  int row = ((blockIdx.x & 127) << 8) + threadIdx.x;
  const float* e = enc + (size_t)t * RH + (size_t)row * HIDD;
  float acc = 0.f;
#pragma unroll
  for (int j = 0; j < 8; ++j) {
    float s = bm1[j];
    const float* w = wm1 + j * HIDD;
    for (int k = 0; k < HIDD; ++k) s += e[k] * w[k];
    float g = 0.5f * s * (1.f + erff(s * 0.70710678118f));
    acc += g * wm2[j];
  }
  acc += bm2[0];
  red[threadIdx.x] = acc; __syncthreads();
  for (int o = 128; o; o >>= 1) {
    if (threadIdx.x < (unsigned)o) red[threadIdx.x] += red[threadIdx.x + o];
    __syncthreads();
  }
  if (threadIdx.x == 0) atomicAdd(result + t, red[0]);
}

__global__ void k_topk(const float* __restrict__ result, int* __restrict__ idx) {
  if (threadIdx.x != 0 || blockIdx.x != 0) return;
  bool taken[SEQN];
  for (int t = 0; t < SEQN; ++t) taken[t] = false;
  for (int k = 0; k < TOPKN; ++k) {
    float best = -3.0e38f; int bi = 0;
    for (int t = 0; t < SEQN; ++t)
      if (!taken[t] && result[t] > best) { best = result[t]; bi = t; }
    taken[bi] = true; idx[k] = bi;
  }
}

// eo = enc_flat[idx].reshape(-1, 9, 64): flat(i*9+j) = t_pos*32768 + r (torch-faithful)
__global__ void k_build_eo(const float* __restrict__ enc, const int* __restrict__ idx,
                           uint16_t* __restrict__ eo) {
  int tid = blockIdx.x * blockDim.x + threadIdx.x;
  if (tid >= ROWS * TOPKN * HIDD) return;
  int p = tid >> 6, k = tid & 63;
  int tpos = p >> 15, r = p & 32767;
  eo[tid] = f2bf(enc[(size_t)idx[tpos] * RH + (size_t)r * HIDD + k]);
}

__global__ void k_f2bf_arr(const float* __restrict__ src, uint16_t* __restrict__ dst, int n) {
  int tid = blockIdx.x * blockDim.x + threadIdx.x;
  if (tid < n) dst[tid] = f2bf(src[tid]);
}

// GRU gates: hd = (1-z)*n + z*h
__global__ void k_gru_gate(const float* __restrict__ gi, const float* __restrict__ gh,
                           const float* __restrict__ hprev, float* __restrict__ hf,
                           uint16_t* __restrict__ hbf) {
  int tid = blockIdx.x * blockDim.x + threadIdx.x;
  if (tid >= RH) return;
  int row = tid >> 6, f = tid & 63;
  const float* a = gi + (size_t)row * 192;
  const float* b = gh + (size_t)row * 192;
  float r  = sigf(a[f] + b[f]);
  float z  = sigf(a[64 + f] + b[64 + f]);
  float nn = tanhf(a[128 + f] + r * b[128 + f]);
  float hd = (1.f - z) * nn + z * hprev[tid];
  hf[tid] = hd; hbf[tid] = f2bf(hd);
}

// Attention + context + prediction + feedback; one wave32 per row
__global__ void k_attn(const float* __restrict__ hf, const float* __restrict__ hdp,
                       const uint16_t* __restrict__ eo, const uint16_t* __restrict__ eop,
                       const float* __restrict__ wv, const float* __restrict__ wll,
                       const float* __restrict__ bll, uint16_t* __restrict__ xbf,
                       float* __restrict__ out, int step) {
  int lane = threadIdx.x & 31;
  int row  = blockIdx.x * 8 + (threadIdx.x >> 5);
  if (row >= ROWS) return;
  int f0 = lane << 1, f1 = f0 | 1;
  float hd0 = hf[(size_t)row * 64 + f0], hd1 = hf[(size_t)row * 64 + f1];
  float hp0 = hdp[(size_t)row * 64 + f0], hp1 = hdp[(size_t)row * 64 + f1];
  float wv0 = wv[f0], wv1 = wv[f1];
  float logit[TOPKN];
#pragma unroll
  for (int s = 0; s < TOPKN; ++s) {
    const uint16_t* ep = eop + (size_t)(row * TOPKN + s) * HIDD;
    float v = tanhf(hp0 + bf2f(ep[f0])) * wv0 + tanhf(hp1 + bf2f(ep[f1])) * wv1;
    for (int o = 16; o; o >>= 1) v += __shfl_xor(v, o);
    logit[s] = v;
  }
  float mx = logit[0];
#pragma unroll
  for (int s = 1; s < TOPKN; ++s) mx = fmaxf(mx, logit[s]);
  float att[TOPKN]; float den = 0.f;
#pragma unroll
  for (int s = 0; s < TOPKN; ++s) { att[s] = expf(logit[s] - mx); den += att[s]; }
  float inv = 1.f / den;
  float c0 = 0.f, c1 = 0.f;
#pragma unroll
  for (int s = 0; s < TOPKN; ++s) {
    const uint16_t* er = eo + (size_t)(row * TOPKN + s) * HIDD;
    float a = att[s] * inv;
    c0 += a * bf2f(er[f0]);
    c1 += a * bf2f(er[f1]);
  }
  xbf[(size_t)row * KPAD + f0] = f2bf(c0);
  xbf[(size_t)row * KPAD + f1] = f2bf(c1);
  float pp = tanhf(c0) * wll[f0] + tanhf(c1) * wll[f1]
           + tanhf(hd0) * wll[64 + f0] + tanhf(hd1) * wll[64 + f1];
  for (int o = 16; o; o >>= 1) pp += __shfl_xor(pp, o);
  if (lane == 0) {
    float pred = pp + bll[0];
    out[(size_t)row * NPREDN + step] = pred;
    xbf[(size_t)row * KPAD + 64] = f2bf(pred);
  }
}

extern "C" void kernel_launch(void* const* d_in, const int* in_sizes, int n_in,
                              void* d_out, int out_size, void* d_ws, size_t ws_size,
                              hipStream_t stream) {
  (void)in_sizes; (void)n_in; (void)out_size; (void)ws_size;
  const float* train  = (const float*)d_in[0];
  const float* adj    = (const float*)d_in[2];
  const float* w_gcn1 = (const float*)d_in[3];
  const float* b_gcn1 = (const float*)d_in[4];
  const float* w_gcn2 = (const float*)d_in[5];
  const float* b_gcn2 = (const float*)d_in[6];
  const float* w_ih   = (const float*)d_in[7];
  const float* w_hh   = (const float*)d_in[8];
  const float* b_ih   = (const float*)d_in[9];
  const float* b_hh   = (const float*)d_in[10];
  const float* w_attn = (const float*)d_in[11];
  const float* w_v    = (const float*)d_in[12];
  const float* wm1    = (const float*)d_in[13];
  const float* bm1    = (const float*)d_in[14];
  const float* wm2    = (const float*)d_in[15];
  const float* bm2    = (const float*)d_in[16];
  const float* w_ll   = (const float*)d_in[17];
  const float* b_ll   = (const float*)d_in[18];
  float* out = (float*)d_out;

  char* p = (char*)d_ws;
  auto alloc = [&](size_t bytes) -> char* {
    char* r = p; p += (bytes + 255) & ~(size_t)255; return r;
  };
  float*    dinv  = (float*)   alloc((size_t)NODES * 4);
  uint16_t* ah    = (uint16_t*)alloc((size_t)NODES * NODES * 2);
  uint16_t* cct   = (uint16_t*)alloc((size_t)COLS * NODES * 2);
  uint16_t* AC    = (uint16_t*)alloc((size_t)ROWS * KPAD * 2);
  float*    sigG  = (float*)   alloc((size_t)ROWS * 128 * 4);
  float*    c2    = (float*)   alloc((size_t)RH * 4);
  float*    h0    = (float*)   alloc((size_t)RH * 4);
  float*    enc   = (float*)   alloc((size_t)SEQN * RH * 4);
  uint16_t* w1t   = (uint16_t*)alloc((size_t)128 * KPAD * 2);
  uint16_t* w2t   = (uint16_t*)alloc((size_t)64 * KPAD * 2);
  uint16_t* wiht  = (uint16_t*)alloc((size_t)192 * KPAD * 2);
  uint16_t* whht  = (uint16_t*)alloc((size_t)192 * 64 * 2);
  uint16_t* waht  = (uint16_t*)alloc((size_t)64 * 64 * 2);
  uint16_t* waet  = (uint16_t*)alloc((size_t)64 * 64 * 2);
  float*    result= (float*)   alloc(SEQN * 4);
  int*      idx   = (int*)     alloc(TOPKN * 4);
  uint16_t* eo    = (uint16_t*)alloc((size_t)ROWS * TOPKN * HIDD * 2);
  uint16_t* eop   = (uint16_t*)alloc((size_t)ROWS * TOPKN * HIDD * 2);
  uint16_t* xbf   = (uint16_t*)alloc((size_t)ROWS * KPAD * 2);
  float*    gi    = (float*)   alloc((size_t)ROWS * 192 * 4);
  float*    gh    = (float*)   alloc((size_t)ROWS * 192 * 4);
  float*    hfb   = (float*)   alloc((size_t)RH * 4);
  uint16_t* hbf   = (uint16_t*)alloc((size_t)RH * 2);
  float*    hdp   = (float*)   alloc((size_t)RH * 4);

  hipMemsetAsync(AC,  0, (size_t)ROWS * KPAD * 2, stream);   // pad cols stay zero
  hipMemsetAsync(xbf, 0, (size_t)ROWS * KPAD * 2, stream);   // x0 = zeros(65) + pad
  hipMemsetAsync(h0,  0, (size_t)RH * 4, stream);
  hipMemsetAsync(result, 0, SEQN * 4, stream);

  // adjacency normalization + weight staging
  k_deg <<<NODES / 8, 256, 0, stream>>>(adj, dinv);
  k_ahat<<<(NODES * NODES + 255) / 256, 256, 0, stream>>>(adj, dinv, ah);
  k_stage_w<<<(128 * KPAD + 255) / 256, 256, 0, stream>>>(w1t,  w_gcn1,     128, F1, KPAD, 1,   128);
  k_stage_w<<<(64  * KPAD + 255) / 256, 256, 0, stream>>>(w2t,  w_gcn2,     64,  F1, KPAD, 1,   64);
  k_stage_w<<<(192 * KPAD + 255) / 256, 256, 0, stream>>>(wiht, w_ih,       192, F1, KPAD, F1,  1);
  k_stage_w<<<(192 * 64   + 255) / 256, 256, 0, stream>>>(whht, w_hh,       192, 64, 64,   64,  1);
  k_stage_w<<<(64  * 64   + 255) / 256, 256, 0, stream>>>(waht, w_attn,     64,  64, 64,   128, 1);
  k_stage_w<<<(64  * 64   + 255) / 256, 256, 0, stream>>>(waet, w_attn + 64,64,  64, 64,   128, 1);

  const int ccBlocks   = (COLS * NODES + 255) / 256;
  const int ahatBlocks = ((NODES / 32) * (COLS / 32) + 7) / 8;     // 260
  const int ewBlocks   = RH / 256;                                 // 8192
  auto rmBlocks = [](int M, int N) { return ((M >> 5) * (N >> 5) + 7) / 8; };

  // -------- encoder: 12 GCN-GRU steps --------
  for (int t = 0; t < SEQN; ++t) {
    const float* hprev = (t == 0) ? h0 : (enc + (size_t)(t - 1) * RH);
    float* hout = enc + (size_t)t * RH;
    // gate GCN: g = sigmoid(gcn1(a_hat, x, h))
    k_build_cc<<<ccBlocks, 256, 0, stream>>>(train, hprev, nullptr, cct, t, 0);
    k_gemm_ahat<<<ahatBlocks, 256, 0, stream>>>(ah, cct, AC);
    k_gemm_rm<<<rmBlocks(ROWS, 128), 256, 0, stream>>>(
        AC, w1t, b_gcn1, sigG, nullptr, ROWS, 128, KPAD, 1);
    // candidate GCN: c = tanh(gcn2(a_hat, x, r*h))
    k_build_cc<<<ccBlocks, 256, 0, stream>>>(train, hprev, sigG, cct, t, 1);
    k_gemm_ahat<<<ahatBlocks, 256, 0, stream>>>(ah, cct, AC);
    k_gemm_rm<<<rmBlocks(ROWS, 64), 256, 0, stream>>>(
        AC, w2t, b_gcn2, c2, nullptr, ROWS, 64, KPAD, 2);
    k_update_h<<<ewBlocks, 256, 0, stream>>>(hprev, sigG, c2, hout);
  }

  // -------- scoring + top-k + eo gather + hoisted eo@waE --------
  k_score<<<SEQN * 128, 256, 0, stream>>>(enc, wm1, bm1, wm2, bm2, result);
  k_topk<<<1, 32, 0, stream>>>(result, idx);
  k_build_eo<<<(ROWS * TOPKN * HIDD + 255) / 256, 256, 0, stream>>>(enc, idx, eo);
  k_gemm_rm<<<rmBlocks(ROWS * TOPKN, 64), 256, 0, stream>>>(
      eo, waet, nullptr, nullptr, eop, ROWS * TOPKN, 64, 64, 0);
  k_f2bf_arr<<<ewBlocks, 256, 0, stream>>>(enc + (size_t)(SEQN - 1) * RH, hbf, RH);

  // -------- decoder: 12 GRU + attention steps --------
  for (int t = 0; t < NPREDN; ++t) {
    const float* hprev = (t == 0) ? (enc + (size_t)(SEQN - 1) * RH) : hfb;
    k_gemm_rm<<<rmBlocks(ROWS, 192), 256, 0, stream>>>(
        xbf, wiht, b_ih, gi, nullptr, ROWS, 192, KPAD, 0);
    k_gemm_rm<<<rmBlocks(ROWS, 192), 256, 0, stream>>>(
        hbf, whht, b_hh, gh, nullptr, ROWS, 192, 64, 0);
    k_gru_gate<<<ewBlocks, 256, 0, stream>>>(gi, gh, hprev, hfb, hbf);
    k_gemm_rm<<<rmBlocks(ROWS, 64), 256, 0, stream>>>(
        hbf, waht, nullptr, hdp, nullptr, ROWS, 64, 64, 0);
    k_attn<<<ROWS / 8, 256, 0, stream>>>(hfb, hdp, eo, eop, w_v, w_ll, b_ll, xbf, out, t);
  }
}
